// FlashAttention_87436944212727
// MI455X (gfx1250) — matrix operand order repaired and checked by reading, not compiled, not run on hardware
//
#include <hip/hip_runtime.h>
#include <hip/hip_bf16.h>

// MI455X / gfx1250, wave32. Mixed precision: one-shot fp32->bf16 conversion
// pass, then all GEMMs run on v_wmma_f32_16x16x32_bf16 with f32 accumulation.
// V tiles are staged to LDS with the CDNA5 async global->LDS path when the
// toolchain exposes it (ASYNCcnt-tracked, overlapped with QK^T WMMAs).

#define EMBED 1024
#define NH    16
#define HD    64
#define BB    2
#define SS    2048
#define M_TOT (BB * SS)   // 4096 rows

typedef __attribute__((ext_vector_type(16))) __bf16 bf16x16;
typedef __attribute__((ext_vector_type(8)))  __bf16 bf16x8;
typedef __attribute__((ext_vector_type(8)))  float  f32x8;

#if __has_builtin(__builtin_amdgcn_global_load_async_to_lds_b128)
#define HAS_ASYNC_LDS 1
// builtin signature (from hipcc diagnostic): param0 = v4i in AS(1) (global),
// param1 = LDS destination -> AS(3), then imm offset + imm cpol.
typedef __attribute__((ext_vector_type(4))) int i32x4_t;
typedef __attribute__((address_space(1))) i32x4_t gvec_t;
typedef __attribute__((address_space(3))) i32x4_t lvec_t;
#else
#define HAS_ASYNC_LDS 0
#endif

static __device__ __forceinline__ f32x8 wmma_bf16(bf16x16 a, bf16x16 b, f32x8 c) {
  return __builtin_amdgcn_wmma_f32_16x16x32_bf16(
      /*neg_a=*/false, a, /*neg_b=*/false, b,
      /*c_mod=*/(short)0, c, /*reuse_a=*/false, /*reuse_b=*/false);
}

// ---- operand loaders, contiguous k order shared by A and B (wave32) ----------------
// A-matrix 16x32 (MxK), 16-bit: lane m = L%16, K = 16*(L/16) + e (contiguous).
static __device__ __forceinline__ bf16x16 load_a_bf16(const __bf16* p, int ld, int lane) {
  const int m  = lane & 15;
  const int kb = (lane >> 4) << 4;
  const __bf16* r0 = p + (size_t)m * ld + kb;
  bf16x16 a;
#pragma unroll
  for (int i = 0; i < 16; ++i) a[i] = r0[i];
  return a;
}

// B-matrix 32x16 (KxN), 16-bit: lane n = L%16, K = 16*(L/16) + e (contiguous).
// Memory holds W[n][k] row-major (B(k,n) = W[n][k]) -> 32B contiguous per lane.
static __device__ __forceinline__ bf16x16 load_bT_bf16(const __bf16* p, int ld, int lane) {
  const int n  = lane & 15;
  const int kb = (lane >> 4) << 4;
  const __bf16* r0 = p + (size_t)n * ld + kb;
  bf16x16 b;
#pragma unroll
  for (int i = 0; i < 16; ++i) b[i] = r0[i];
  return b;
}

// ---- kernel 0: bulk fp32 -> bf16 conversion (bandwidth-bound, one pass) ----
__global__ __launch_bounds__(256) void cvt_f32_bf16(
    const float* __restrict__ src, __bf16* __restrict__ dst, int n8) {
  const int i = blockIdx.x * blockDim.x + threadIdx.x;
  if (i >= n8) return;
  const float4* s = (const float4*)src + (size_t)i * 2;
  const float4 a = s[0], b = s[1];
  bf16x8 o;
  o[0] = (__bf16)a.x; o[1] = (__bf16)a.y; o[2] = (__bf16)a.z; o[3] = (__bf16)a.w;
  o[4] = (__bf16)b.x; o[5] = (__bf16)b.y; o[6] = (__bf16)b.z; o[7] = (__bf16)b.w;
  *(bf16x8*)(dst + (size_t)i * 8) = o;
}

// ---- kernel 1: fused QKV projection, scatter to [B][H][S][D] bf16 ----------
// One wave computes a 16(M) x 128(N) tile: 1 A-load amortized over 8 WMMAs.
__global__ __launch_bounds__(32) void qkv_gemm(
    const __bf16* __restrict__ xb, const __bf16* __restrict__ wb,
    const float* __restrict__ bias,
    __bf16* __restrict__ qB, __bf16* __restrict__ kB, __bf16* __restrict__ vB) {
  const int lane = threadIdx.x;
  const int m0 = blockIdx.x * 16;    // row tile in [0, 4096)
  const int n0 = blockIdx.y * 128;   // col tile in [0, 3072)
  f32x8 acc[8] = {};
  for (int kt = 0; kt < EMBED; kt += 32) {
    const bf16x16 a = load_a_bf16(xb + (size_t)m0 * EMBED + kt, EMBED, lane);
#pragma unroll
    for (int j = 0; j < 8; ++j) {
      const bf16x16 b = load_bT_bf16(wb + (size_t)(n0 + j * 16) * EMBED + kt, EMBED, lane);
      acc[j] = wmma_bf16(a, b, acc[j]);
    }
  }
  // epilogue: C layout lane holds (m = r + 8*(L/16), n = L%16)
#pragma unroll
  for (int j = 0; j < 8; ++j) {
    const int n = n0 + j * 16 + (lane & 15);
    const float bv = bias[n];
    const int which = n >> 10;          // 0=Q 1=K 2=V
    const int e = n & 1023;
    const int h = e >> 6, d = e & 63;
    __bf16* dst = (which == 0) ? qB : (which == 1) ? kB : vB;
#pragma unroll
    for (int r = 0; r < 8; ++r) {
      const int m = m0 + r + ((lane >> 4) << 3);
      const int bi = m >> 11, s = m & (SS - 1);
      dst[((size_t)(bi * NH + h) * SS + s) * HD + d] = (__bf16)(acc[j][r] + bv);
    }
  }
}

// ---- kernel 2: flash attention, one wave per (b,h,16-query) tile -----------
__global__ __launch_bounds__(32) void flash_attn(
    const __bf16* __restrict__ qB, const __bf16* __restrict__ kB,
    const __bf16* __restrict__ vB, const int* __restrict__ mask,
    __bf16* __restrict__ attnB) {
  __shared__ __bf16 ldsV[32 * HD];   // 32 keys x 64 dims staged tile
  __shared__ __bf16 ldsP[16 * 32];   // probabilities, re-read in A layout

  const int lane = threadIdx.x;
  const int qt = blockIdx.x * 16;
  const int bh = blockIdx.y;               // b*NH + h
  const int bi = bh >> 4, h = bh & 15;

  const __bf16* qbase = qB + ((size_t)bh * SS + qt) * HD;
  const __bf16* kbase = kB + (size_t)bh * SS * HD;
  const __bf16* vbase = vB + (size_t)bh * SS * HD;
  const int* mrow = mask + (size_t)bi * SS;

  // Q tile 16x64 resident in registers as two A operands (dims 0..31, 32..63)
  const bf16x16 aq0 = load_a_bf16(qbase, HD, lane);
  const bf16x16 aq1 = load_a_bf16(qbase + 32, HD, lane);

  f32x8 o[4] = {};
  float mrun[8], lrun[8];
#pragma unroll
  for (int r = 0; r < 8; ++r) { mrun[r] = -3.0e38f; lrun[r] = 0.0f; }

  const float scale = 0.125f;   // 1/sqrt(64)

  for (int kt = 0; kt < SS; kt += 32) {
    // speculative prefetch of the next K/V tiles (global_prefetch_b8)
    if (kt + 32 < SS) {
      __builtin_prefetch(kbase + (size_t)(kt + 32) * HD + lane * 4, 0, 0);
      __builtin_prefetch(vbase + (size_t)(kt + 32) * HD + lane * 4, 0, 0);
    }

    // stage V tile (32x64 bf16 = 4KB): lane owns one 128B row.
    // Prefer the CDNA5 async global->LDS path: copies bypass VGPRs, are
    // tracked by ASYNCcnt, and overlap with the QK^T WMMAs issued below.
#if HAS_ASYNC_LDS
    {
      const __bf16* srcRow = vbase + (size_t)(kt + lane) * HD;
      __bf16* dstRow = ldsV + lane * HD;
#pragma unroll
      for (int i = 0; i < 8; ++i)
        __builtin_amdgcn_global_load_async_to_lds_b128(
            (gvec_t*)(srcRow + i * 8), (lvec_t*)(dstRow + i * 8), 0, 0);
    }
#else
    {
      const uint4* src = (const uint4*)(vbase + (size_t)(kt + lane) * HD);
      uint4* dst = (uint4*)(ldsV + lane * HD);
#pragma unroll
      for (int i = 0; i < 8; ++i) dst[i] = src[i];
    }
#endif

    // S = Q K^T for keys kt..kt+15 (s0) and kt+16..kt+31 (s1)
    // (V async-copy is in flight during these WMMAs and the softmax)
    f32x8 s0 = {}, s1 = {};
    {
      const bf16x16 b00 = load_bT_bf16(kbase + (size_t)kt * HD, HD, lane);
      const bf16x16 b01 = load_bT_bf16(kbase + (size_t)kt * HD + 32, HD, lane);
      s0 = wmma_bf16(aq0, b00, s0);
      s0 = wmma_bf16(aq1, b01, s0);
      const bf16x16 b10 = load_bT_bf16(kbase + (size_t)(kt + 16) * HD, HD, lane);
      const bf16x16 b11 = load_bT_bf16(kbase + (size_t)(kt + 16) * HD + 32, HD, lane);
      s1 = wmma_bf16(aq0, b10, s1);
      s1 = wmma_bf16(aq1, b11, s1);
    }

    // key-padding mask + online softmax (rows distributed over 16-lane halves)
    const int mk0 = mrow[kt + (lane & 15)];
    const int mk1 = mrow[kt + 16 + (lane & 15)];
    float p0[8], p1[8];
#pragma unroll
    for (int r = 0; r < 8; ++r) {
      const float v0 = mk0 ? s0[r] * scale : -3.0e38f;
      const float v1 = mk1 ? s1[r] * scale : -3.0e38f;
      float tm = fmaxf(v0, v1);
      tm = fmaxf(tm, __shfl_xor(tm, 1, 32));
      tm = fmaxf(tm, __shfl_xor(tm, 2, 32));
      tm = fmaxf(tm, __shfl_xor(tm, 4, 32));
      tm = fmaxf(tm, __shfl_xor(tm, 8, 32));
      const float mn = fmaxf(mrun[r], tm);
      const float sc = __expf(mrun[r] - mn);
      const float e0 = mk0 ? __expf(v0 - mn) : 0.0f;
      const float e1 = mk1 ? __expf(v1 - mn) : 0.0f;
      float rs = e0 + e1;
      rs += __shfl_xor(rs, 1, 32);
      rs += __shfl_xor(rs, 2, 32);
      rs += __shfl_xor(rs, 4, 32);
      rs += __shfl_xor(rs, 8, 32);
      lrun[r] = lrun[r] * sc + rs;
      mrun[r] = mn;
#pragma unroll
      for (int j = 0; j < 4; ++j) o[j][r] *= sc;
      p0[r] = e0;
      p1[r] = e1;
    }

    // cross-lane transpose of P (C layout -> A layout) through LDS
#pragma unroll
    for (int r = 0; r < 8; ++r) {
      const int m = r + ((lane >> 4) << 3);
      ldsP[m * 32 + (lane & 15)]      = (__bf16)p0[r];
      ldsP[m * 32 + 16 + (lane & 15)] = (__bf16)p1[r];
    }

    // drain the async V copy, then make all LDS writes visible wave-wide
#if HAS_ASYNC_LDS
#if __has_builtin(__builtin_amdgcn_s_wait_asynccnt)
    __builtin_amdgcn_s_wait_asynccnt(0);
#else
    asm volatile("s_wait_asynccnt 0" ::: "memory");
#endif
#endif
    __syncthreads();

    const bf16x16 pa = load_a_bf16(ldsP, 32, lane);

    // O += P @ V  (V tile 32x64 as four 32x16 B operands from LDS)
#pragma unroll
    for (int j = 0; j < 4; ++j) {
      const int n  = j * 16 + (lane & 15);
      const int kb = (lane >> 4) << 4;
      bf16x16 bv;
#pragma unroll
      for (int e = 0; e < 16; ++e) bv[e] = ldsV[(kb + e) * HD + n];
      o[j] = wmma_bf16(pa, bv, o[j]);
    }
    __syncthreads();   // before next iteration reuses ldsV/ldsP
  }

  // normalize and write attn as bf16 in [b][s][h*64+d] row-major layout
#pragma unroll
  for (int j = 0; j < 4; ++j) {
    const int d = j * 16 + (lane & 15);
#pragma unroll
    for (int r = 0; r < 8; ++r) {
      const int m = r + ((lane >> 4) << 3);
      const int s = qt + m;
      const float l = lrun[r];
      const float val = (l > 0.0f) ? o[j][r] / l : 0.0f;
      attnB[((size_t)(bi * SS + s) * NH + h) * HD + d] = (__bf16)val;
    }
  }
}

// ---- kernel 3: output projection, 16x128 tile per wave ---------------------
__global__ __launch_bounds__(32) void out_gemm(
    const __bf16* __restrict__ attnB, const __bf16* __restrict__ wb,
    const float* __restrict__ bias, float* __restrict__ out) {
  const int lane = threadIdx.x;
  const int m0 = blockIdx.x * 16;
  const int n0 = blockIdx.y * 128;
  f32x8 acc[8] = {};
  for (int kt = 0; kt < EMBED; kt += 32) {
    const bf16x16 a = load_a_bf16(attnB + (size_t)m0 * EMBED + kt, EMBED, lane);
#pragma unroll
    for (int j = 0; j < 8; ++j) {
      const bf16x16 b = load_bT_bf16(wb + (size_t)(n0 + j * 16) * EMBED + kt, EMBED, lane);
      acc[j] = wmma_bf16(a, b, acc[j]);
    }
  }
#pragma unroll
  for (int j = 0; j < 8; ++j) {
    const int n = n0 + j * 16 + (lane & 15);
    const float bv = bias[n];
#pragma unroll
    for (int r = 0; r < 8; ++r) {
      const int m = m0 + r + ((lane >> 4) << 3);
      out[(size_t)m * EMBED + n] = acc[j][r] + bv;
    }
  }
}

extern "C" void kernel_launch(void* const* d_in, const int* in_sizes, int n_in,
                              void* d_out, int out_size, void* d_ws, size_t ws_size,
                              hipStream_t stream) {
  (void)in_sizes; (void)n_in; (void)out_size; (void)ws_size;
  const float* x     = (const float*)d_in[0];
  const int*   mask  = (const int*)d_in[1];
  const float* qkv_w = (const float*)d_in[2];
  const float* qkv_b = (const float*)d_in[3];
  const float* o_w   = (const float*)d_in[4];
  const float* o_b   = (const float*)d_in[5];
  float* out = (float*)d_out;

  // workspace layout (bf16 elements):
  //   xb 4M | qkvw_b 3M | ow_b 1M | Q 4M | K 4M | V 4M | attn 4M = 48 MB
  __bf16* xb    = (__bf16*)d_ws;
  __bf16* wqkvb = xb    + (size_t)M_TOT * EMBED;
  __bf16* owb   = wqkvb + (size_t)3 * EMBED * EMBED;
  __bf16* qB    = owb   + (size_t)EMBED * EMBED;
  const size_t perBuf = (size_t)BB * NH * SS * HD;   // 4M elements
  __bf16* kB    = qB + perBuf;
  __bf16* vB    = kB + perBuf;
  __bf16* attnB = vB + perBuf;

  // one-shot conversions (bandwidth-bound)
  {
    const int nx = (M_TOT * EMBED) / 8;             // 512K
    const int nw = (3 * EMBED * EMBED) / 8;         // 384K
    const int no = (EMBED * EMBED) / 8;             // 128K
    cvt_f32_bf16<<<(nx + 255) / 256, 256, 0, stream>>>(x, xb, nx);
    cvt_f32_bf16<<<(nw + 255) / 256, 256, 0, stream>>>(qkv_w, wqkvb, nw);
    cvt_f32_bf16<<<(no + 255) / 256, 256, 0, stream>>>(o_w, owb, no);
  }

  qkv_gemm <<<dim3(M_TOT / 16, (3 * EMBED) / 128), 32, 0, stream>>>(xb, wqkvb, qkv_b, qB, kB, vB);
  flash_attn<<<dim3(SS / 16, BB * NH),             32, 0, stream>>>(qB, kB, vB, mask, attnB);
  out_gemm  <<<dim3(M_TOT / 16, EMBED / 128),      32, 0, stream>>>(attnB, owb, o_b, out);
}
